// MultiHeadAttention_32744830665036
// MI455X (gfx1250) — matrix operand order repaired and checked by reading, not compiled, not run on hardware
//
#include <hip/hip_runtime.h>

// ---------------------------------------------------------------------------
// Multi-head attention (B=4, T=2048, C=768, H=12, D=64) for gfx1250 (MI455X).
// f16 WMMA (v_wmma_f32_16x16x32_f16) with f32 accumulation everywhere.
// Round 2: widened wave tiles (GEMM: 32x64/wave, attention: 32 q-rows/wave)
// for ~2x matrix work per LDS fragment load.
// ---------------------------------------------------------------------------

typedef _Float16 v16h __attribute__((ext_vector_type(16)));
typedef _Float16 v8h  __attribute__((ext_vector_type(8)));
typedef float    v8f  __attribute__((ext_vector_type(8)));

#define DEVINL static __device__ __forceinline__

constexpr int BB  = 4;      // batch
constexpr int TT  = 2048;   // sequence
constexpr int CC  = 768;    // channels
constexpr int HH  = 12;     // heads
constexpr int DD  = 64;     // head dim
constexpr int N3C = 3 * CC; // 2304

DEVINL v8f wmma16x16x32(v16h a, v16h b, v8f c) {
  // D = A(16x32) * B(32x16) + C, f16 inputs, f32 accumulate
  return __builtin_amdgcn_wmma_f32_16x16x32_f16(
      /*neg_a=*/false, a, /*neg_b=*/false, b,
      /*c_mod=*/(short)0, c, /*reuse_a=*/false, /*reuse_b=*/false);
}

// A fragment: 16x32 tile, row-major source, row stride ld (halves).
// Order used here (matches B): lane L holds row M=L&15; K = (L>>4)*16 + e, contiguous.
DEVINL v16h load_a16x32(const _Float16* base, int ld, int lane) {
  const int row = lane & 15;
  const int khb = (lane >> 4) * 16;
  const _Float16* p = base + row * ld + khb;
  return *reinterpret_cast<const v16h*>(p);
}

// B fragment: 32x16 tile stored TRANSPOSED as [n][k], n-stride ld (halves).
// Order used here (matches A): lane L holds column N=L&15; K = (L>>4)*16 + e, contiguous.
DEVINL v16h load_bT32x16(const _Float16* base, int ld, int lane) {
  const int n  = lane & 15;
  const int kb = (lane >> 4) * 16;
  return *reinterpret_cast<const v16h*>(base + n * ld + kb);
}

// ---------------------------------------------------------------------------
// f32 -> f16 pack
// ---------------------------------------------------------------------------
__global__ void mha_cvt_f16(const float* __restrict__ s,
                            _Float16* __restrict__ d, int n) {
  for (int i = blockIdx.x * blockDim.x + threadIdx.x; i < n;
       i += gridDim.x * blockDim.x)
    d[i] = (_Float16)s[i];
}

// ---------------------------------------------------------------------------
// Shared GEMM block body: block tile 128(M) x 128(N), 8 waves arranged 4Mx2N,
// each wave computes 32x64 (8 WMMAs per 32-deep k-step).
// As: [128][32] row-major; Bt: [128][32] transposed ([n][k]).
// ---------------------------------------------------------------------------
struct GemmAcc {
  v8f a[2][4];
};

template <int LDW>
DEVINL void gemm_tile_128x128(const _Float16* __restrict__ xh,
                              const _Float16* __restrict__ wh,
                              _Float16* As, _Float16* Bt, GemmAcc& g,
                              int m0, int n0, int lane, int wm, int wn,
                              int tid) {
  for (int kc = 0; kc < CC; kc += 32) {
    __syncthreads();
    {  // stage A: 128 rows x 32 halves, 2 threads/row, b128x2 each
      const int row = tid >> 1;
      const int off = (tid & 1) * 16;
      *reinterpret_cast<v16h*>(&As[row * 32 + off]) =
          *reinterpret_cast<const v16h*>(&xh[(size_t)(m0 + row) * CC + kc + off]);
    }
    {  // stage B transposed: 32 k-rows x 128 n-cols, 8 threads/row x 16 halves
      const int r = tid >> 3;
      const int c = (tid & 7) * 16;
      v16h wv = *reinterpret_cast<const v16h*>(&wh[(size_t)(kc + r) * LDW + n0 + c]);
#pragma unroll
      for (int j = 0; j < 16; ++j) Bt[(c + j) * 32 + r] = wv[j];
    }
    __syncthreads();

    v16h a0 = load_a16x32(&As[(wm * 32 + 0)  * 32], 32, lane);
    v16h a1 = load_a16x32(&As[(wm * 32 + 16) * 32], 32, lane);
#pragma unroll
    for (int ni = 0; ni < 4; ++ni) {
      v16h bf = load_bT32x16(&Bt[(wn * 64 + ni * 16) * 32], 32, lane);
      g.a[0][ni] = wmma16x16x32(a0, bf, g.a[0][ni]);
      g.a[1][ni] = wmma16x16x32(a1, bf, g.a[1][ni]);
    }
  }
}

// ---------------------------------------------------------------------------
// QKV projection: qkv = x @ W_attn + b_attn, scattered into [B,H,T,D] f16.
// ---------------------------------------------------------------------------
__global__ __launch_bounds__(256) void mha_qkv_gemm(
    const _Float16* __restrict__ xh, const _Float16* __restrict__ wh,
    const float* __restrict__ bias, _Float16* __restrict__ Qh,
    _Float16* __restrict__ Kh, _Float16* __restrict__ Vh) {
  const int tid  = threadIdx.x;
  const int lane = tid & 31;
  const int wid  = tid >> 5;
  const int wm   = wid & 3;   // M offset wm*32
  const int wn   = wid >> 2;  // N offset wn*64
  const int m0   = blockIdx.x * 128;
  const int n0   = blockIdx.y * 128;

  __shared__ alignas(32) _Float16 As[128 * 32];
  __shared__ alignas(32) _Float16 Bt[128 * 32];

  GemmAcc g = {};
  gemm_tile_128x128<N3C>(xh, wh, As, Bt, g, m0, n0, lane, wm, wn, tid);

  // epilogue: bias + scatter into Q/K/V [B,H,T,D] f16
  const int mro = (lane >> 4) * 8;
  const int nl  = lane & 15;
#pragma unroll
  for (int mi = 0; mi < 2; ++mi)
#pragma unroll
    for (int ni = 0; ni < 4; ++ni) {
      const int col = n0 + wn * 64 + ni * 16 + nl;  // 0..2303
      const float bv = bias[col];
      const int sec = col / CC;                     // 0=Q 1=K 2=V
      const int cc  = col - sec * CC;
      const int h   = cc >> 6;
      const int d   = cc & 63;
      _Float16* dst = (sec == 0) ? Qh : ((sec == 1) ? Kh : Vh);
      const int rbase = m0 + wm * 32 + mi * 16 + mro;
#pragma unroll
      for (int r = 0; r < 8; ++r) {
        const int row = rbase + r;            // 0..8191 = b*T + t
        const int b   = row >> 11;
        const int t   = row & (TT - 1);
        dst[(((size_t)b * HH + h) * TT + t) * DD + d] =
            (_Float16)(g.a[mi][ni][r] + bv);
      }
    }
}

// ---------------------------------------------------------------------------
// Flash attention: one (b,h) x 128 query rows per block, 4 waves x 32 rows.
// Keys stream in steps of 32; K and V fragments are loaded once per step and
// reused across both 16-row subtiles -> 16 WMMAs per 8 fragment loads.
// ---------------------------------------------------------------------------
__global__ __launch_bounds__(128) void mha_flash_attn(
    const _Float16* __restrict__ Qh, const _Float16* __restrict__ Kh,
    const _Float16* __restrict__ Vh, _Float16* __restrict__ Yh) {
  const int lane = threadIdx.x & 31;
  const int wid  = threadIdx.x >> 5;
  const int q0   = blockIdx.x * 128;
  const int bh   = blockIdx.y;  // b*H + h
  const int b    = bh / HH;
  const int h    = bh - b * HH;

  __shared__ alignas(32) _Float16 Vt[DD * 32];      // [d][key] transposed
  __shared__ alignas(32) _Float16 Pb[4][16 * 32];   // per-wave P tile

  const _Float16* Qhead = Qh + (size_t)bh * TT * DD;
  const _Float16* Khead = Kh + (size_t)bh * TT * DD;
  const _Float16* Vhead = Vh + (size_t)bh * TT * DD;

  // Q fragments: 32 rows/wave = two 16-row subtiles x two 16x32 A-frags
  v16h qa[2][2];
#pragma unroll
  for (int u = 0; u < 2; ++u)
#pragma unroll
    for (int dsp = 0; dsp < 2; ++dsp)
      qa[u][dsp] = load_a16x32(
          Qhead + (size_t)(q0 + wid * 32 + u * 16) * DD + dsp * 32, DD, lane);

  const float NEG = -__builtin_inff();
  v8f acc[2][4] = {};
  float mrun[2][8], lrun[2][8];
#pragma unroll
  for (int u = 0; u < 2; ++u)
#pragma unroll
    for (int r = 0; r < 8; ++r) { mrun[u][r] = NEG; lrun[u][r] = 0.f; }

  const int mro = (lane >> 4) * 8;
  const int nl  = lane & 15;
  _Float16* Pw = &Pb[wid][0];

  const int kend = q0 + 128;  // causal: keys <= last query row of block
  for (int kb = 0; kb < kend; kb += 32) {
    __syncthreads();
    {  // stage V tile [32 keys x 64 d] transposed -> Vt[d][key]
      const int key  = threadIdx.x >> 2;
      const int doff = (threadIdx.x & 3) * 16;
      v16h vv = *reinterpret_cast<const v16h*>(
          Vhead + (size_t)(kb + key) * DD + doff);
#pragma unroll
      for (int j = 0; j < 16; ++j) Vt[(doff + j) * 32 + key] = vv[j];
    }
    __syncthreads();

    if (kb + 32 < kend) {  // hint next K/V tiles toward the caches
      __builtin_prefetch(Khead + (size_t)(kb + 32) * DD, 0, 0);
      __builtin_prefetch(Vhead + (size_t)(kb + 32) * DD, 0, 0);
    }

    // K fragments (rows contiguous in d == WMMA B layout -> direct global)
    v16h kf[2][2];  // [key-subtile][d-step]
#pragma unroll
    for (int kt = 0; kt < 2; ++kt)
#pragma unroll
      for (int dsp = 0; dsp < 2; ++dsp)
        kf[kt][dsp] = load_bT32x16(
            Khead + (size_t)(kb + kt * 16) * DD + dsp * 32, DD, lane);
    // V fragments from LDS, reused by both row-subtiles
    v16h vf[4];
#pragma unroll
    for (int i = 0; i < 4; ++i)
      vf[i] = load_bT32x16(&Vt[(i * 16) * 32], 32, lane);

#pragma unroll
    for (int u = 0; u < 2; ++u) {
      // S = Q K^T for two 16-key tiles
      v8f s0 = {}, s1 = {};
#pragma unroll
      for (int dsp = 0; dsp < 2; ++dsp) {
        s0 = wmma16x16x32(qa[u][dsp], kf[0][dsp], s0);
        s1 = wmma16x16x32(qa[u][dsp], kf[1][dsp], s1);
      }

      // streaming softmax over these 32 keys
#pragma unroll
      for (int r = 0; r < 8; ++r) {
        const int qr = q0 + wid * 32 + u * 16 + mro + r;
        const int c0 = kb + nl;
        const int c1 = c0 + 16;
        const bool ok0 = (c0 <= qr), ok1 = (c1 <= qr);
        const float a0 = ok0 ? s0[r] * 0.125f : NEG;  // 1/sqrt(64)
        const float a1 = ok1 ? s1[r] * 0.125f : NEG;
        float tmax = fmaxf(a0, a1);
        tmax = fmaxf(tmax, __shfl_xor(tmax, 1));
        tmax = fmaxf(tmax, __shfl_xor(tmax, 2));
        tmax = fmaxf(tmax, __shfl_xor(tmax, 4));
        tmax = fmaxf(tmax, __shfl_xor(tmax, 8));
        const float mn = fmaxf(mrun[u][r], tmax);
        const float sc = (mn == NEG) ? 1.f : __expf(mrun[u][r] - mn);
        const float p0 = ok0 ? __expf(a0 - mn) : 0.f;
        const float p1 = ok1 ? __expf(a1 - mn) : 0.f;
        float rs = p0 + p1;
        rs += __shfl_xor(rs, 1);
        rs += __shfl_xor(rs, 2);
        rs += __shfl_xor(rs, 4);
        rs += __shfl_xor(rs, 8);
        lrun[u][r] = lrun[u][r] * sc + rs;
        mrun[u][r] = mn;
#pragma unroll
        for (int i = 0; i < 4; ++i) acc[u][i][r] *= sc;
        Pw[(mro + r) * 32 + nl]      = (_Float16)p0;
        Pw[(mro + r) * 32 + 16 + nl] = (_Float16)p1;
      }

      // P(16x32, via per-wave LDS relayout) @ V(32x64): 4 WMMAs
      const v16h pf = load_a16x32(Pw, 32, lane);
#pragma unroll
      for (int i = 0; i < 4; ++i)
        acc[u][i] = wmma16x16x32(pf, vf[i], acc[u][i]);
    }
  }

  // normalize and write y[b, t, h*64+d] as f16
#pragma unroll
  for (int u = 0; u < 2; ++u)
#pragma unroll
    for (int r = 0; r < 8; ++r) {
      const float inv = 1.f / lrun[u][r];
      const int t = q0 + wid * 32 + u * 16 + mro + r;
      const size_t orow = ((size_t)b * TT + t) * CC;
#pragma unroll
      for (int i = 0; i < 4; ++i)
        Yh[orow + h * 64 + i * 16 + nl] = (_Float16)(acc[u][i][r] * inv);
    }
}

// ---------------------------------------------------------------------------
// Output projection: out = y @ W_proj + b_proj  (f32 result)
// ---------------------------------------------------------------------------
__global__ __launch_bounds__(256) void mha_proj_gemm(
    const _Float16* __restrict__ yh, const _Float16* __restrict__ wh,
    const float* __restrict__ bias, float* __restrict__ out) {
  const int tid  = threadIdx.x;
  const int lane = tid & 31;
  const int wid  = tid >> 5;
  const int wm   = wid & 3;
  const int wn   = wid >> 2;
  const int m0   = blockIdx.x * 128;
  const int n0   = blockIdx.y * 128;

  __shared__ alignas(32) _Float16 As[128 * 32];
  __shared__ alignas(32) _Float16 Bt[128 * 32];

  GemmAcc g = {};
  gemm_tile_128x128<CC>(yh, wh, As, Bt, g, m0, n0, lane, wm, wn, tid);

  const int mro = (lane >> 4) * 8;
  const int nl  = lane & 15;
#pragma unroll
  for (int mi = 0; mi < 2; ++mi)
#pragma unroll
    for (int ni = 0; ni < 4; ++ni) {
      const int col = n0 + wn * 64 + ni * 16 + nl;
      const float bv = bias[col];
      const int rbase = m0 + wm * 32 + mi * 16 + mro;
#pragma unroll
      for (int r = 0; r < 8; ++r)
        out[(size_t)(rbase + r) * CC + col] = g.a[mi][ni][r] + bv;
    }
}

// ---------------------------------------------------------------------------
extern "C" void kernel_launch(void* const* d_in, const int* in_sizes, int n_in,
                              void* d_out, int out_size, void* d_ws,
                              size_t ws_size, hipStream_t stream) {
  (void)in_sizes; (void)n_in; (void)out_size; (void)ws_size;
  const float* x      = (const float*)d_in[0];
  const float* W_attn = (const float*)d_in[1];
  const float* b_attn = (const float*)d_in[2];
  const float* W_proj = (const float*)d_in[3];
  const float* b_proj = (const float*)d_in[4];

  // carve workspace (all offsets 256B aligned)
  char* ws = (char*)d_ws;
  size_t off = 0;
  auto carve = [&](size_t bytes) -> char* {
    char* p = ws + off;
    off += (bytes + 255) & ~(size_t)255;
    return p;
  };
  const size_t nBT_C = (size_t)BB * TT * CC;  // 6291456
  _Float16* xh  = (_Float16*)carve(nBT_C * 2);
  _Float16* wah = (_Float16*)carve((size_t)CC * N3C * 2);
  _Float16* wph = (_Float16*)carve((size_t)CC * CC * 2);
  _Float16* Qh  = (_Float16*)carve(nBT_C * 2);
  _Float16* Kh  = (_Float16*)carve(nBT_C * 2);
  _Float16* Vh  = (_Float16*)carve(nBT_C * 2);
  _Float16* Yh  = (_Float16*)carve(nBT_C * 2);

  // 1) pack to f16
  mha_cvt_f16<<<1024, 256, 0, stream>>>(x, xh, (int)nBT_C);
  mha_cvt_f16<<<1024, 256, 0, stream>>>(W_attn, wah, CC * N3C);
  mha_cvt_f16<<<512, 256, 0, stream>>>(W_proj, wph, CC * CC);

  // 2) QKV projection -> Q/K/V [B,H,T,D] f16
  mha_qkv_gemm<<<dim3((BB * TT) / 128, N3C / 128), 256, 0, stream>>>(
      xh, wah, b_attn, Qh, Kh, Vh);

  // 3) causal flash attention -> Yh [B,T,C] f16
  mha_flash_attn<<<dim3(TT / 128, BB * HH), 128, 0, stream>>>(Qh, Kh, Vh, Yh);

  // 4) output projection -> d_out f32
  mha_proj_gemm<<<dim3((BB * TT) / 128, CC / 128), 256, 0, stream>>>(
      Yh, wph, b_proj, (float*)d_out);
}
